// SolutionCalculatorSurface_14456859919015
// MI455X (gfx1250) — compile-verified, matrix operand order checked by reading, not run on hardware
//
#include <hip/hip_runtime.h>

// ---------------------------------------------------------------------------
// Types for CDNA5 WMMA (wave32): D(f32 16x16) = A(bf16 16x32) x B(bf16 32x16) + C
// ---------------------------------------------------------------------------
typedef __attribute__((ext_vector_type(16))) __bf16 v16bf;
typedef __attribute__((ext_vector_type(8)))  __bf16 v8bf;
typedef __attribute__((ext_vector_type(8)))  float  v8f;

#define NV      4       // V variables
#define TP      16      // points per block tile
#define ROWS    112     // 7 rows (center + 6 neighbors) per point * TP

// LDS layout (bytes) -- total 301184 <= 320KB per workgroup
#define OFF_W1     0        // float [4*7*256]            = 28672 B
#define OFF_ROWS   28672    // float [112*8]              =  3584 B
#define OFF_INV    32256    // float [16*6]               =   384 B
#define OFF_RES    32640    // float [4*16]               =   256 B
#define OFF_ENC    32896    // bf16  [16*192]             =  6144 B
#define OFF_ENCPRE 39040    // bf16  [4*16*256]           = 32768 B
#define OFF_HID    71808    // bf16  [4*112*256]          = 229376 B
#define LDS_TOTAL  301184

// packed weight region sizes (halves)
#define NW2    (4*8*8*32*16)    // 131072
#define NA1B   (4*16*4*32*16)   // 131072
#define NA1NG  (4*16*6*32*16)   // 196608

// K-half ordering inside a 16x16x32 bf16 fragment (ISA 7.12.2):
// lanes 0-15 hold K {0..7, 16..23}; lanes 16-31 hold K {8..15, 24..31}
__device__ __forceinline__ int khalf(int lane, int h) {
    int b = (h < 8) ? h : h + 8;
    return (lane < 16) ? b : b + 8;
}

__device__ __forceinline__ v16bf load_frag(const __bf16* p_lo, const __bf16* p_hi) {
    v8bf lo = *(const v8bf*)p_lo;   // ds_load_b128
    v8bf hi = *(const v8bf*)p_hi;   // ds_load_b128
    v16bf r;
#pragma unroll
    for (int i = 0; i < 8; ++i) { r[i] = lo[i]; r[i + 8] = hi[i]; }
    return r;
}

// ---------------------------------------------------------------------------
// Kernel 0: pack fp32 weights into bf16 B-fragment order (contiguous 32B/lane)
// ---------------------------------------------------------------------------
__global__ void pack_weights(const float* __restrict__ W2, const float* __restrict__ A1,
                             unsigned short* __restrict__ pW2u,
                             unsigned short* __restrict__ pA1bu,
                             unsigned short* __restrict__ pA1ngu) {
    __bf16* pW2  = (__bf16*)pW2u;
    __bf16* pA1b = (__bf16*)pA1bu;
    __bf16* pA1ng = (__bf16*)pA1ngu;
    int t = blockIdx.x * blockDim.x + threadIdx.x;
    if (t < NW2) {
        int h = t & 15, lane = (t >> 4) & 31, kc = (t >> 9) & 7, nt = (t >> 12) & 7, v = t >> 15;
        int k = kc * 32 + khalf(lane, h);
        pW2[t] = (__bf16)W2[(v * 256 + k) * 128 + nt * 16 + (lane & 15)];
    } else if (t < NW2 + NA1B) {
        int u = t - NW2;
        int h = u & 15, lane = (u >> 4) & 31, kc = (u >> 9) & 3, nt = (u >> 11) & 15, v = u >> 15;
        int k = kc * 32 + khalf(lane, h);                    // A1 rows 0..127  (A1b)
        pA1b[u] = (__bf16)A1[(v * 320 + k) * 256 + nt * 16 + (lane & 15)];
    } else if (t < NW2 + NA1B + NA1NG) {
        int u = t - NW2 - NA1B;
        int h = u & 15, lane = (u >> 4) & 31;
        int kc = (u >> 9) % 6, nt = (u / 3072) & 15, v = u / 49152;
        int k = kc * 32 + khalf(lane, h);                    // A1 rows 128..319 (A1n|A1g)
        pA1ng[u] = (__bf16)A1[(v * 320 + 128 + k) * 256 + nt * 16 + (lane & 15)];
    }
}

// ---------------------------------------------------------------------------
// Kernel 1: fused per-tile pipeline. 128 threads = 4 waves; wave w owns v = w.
// ---------------------------------------------------------------------------
__global__ __launch_bounds__(128) void fused_kernel(
    const float* __restrict__ centers, const float* __restrict__ enc_g,
    const float* __restrict__ enc_node, const float* __restrict__ nbrs,
    const float* __restrict__ normals, const float* __restrict__ nb_normals,
    const float* __restrict__ areas, const float* __restrict__ nb_areas,
    const float* __restrict__ W1, const float* __restrict__ b1,
    const float* __restrict__ b2, const float* __restrict__ Ab1,
    const float* __restrict__ A2, const float* __restrict__ Ab2,
    const unsigned short* __restrict__ pW2u, const unsigned short* __restrict__ pA1bu,
    const unsigned short* __restrict__ pA1ngu, float* __restrict__ out) {
    extern __shared__ char smem[];
    float*  W1s     = (float*)(smem + OFF_W1);
    float*  rowsS   = (float*)(smem + OFF_ROWS);
    float*  invS    = (float*)(smem + OFF_INV);
    float*  resS    = (float*)(smem + OFF_RES);
    __bf16* encS    = (__bf16*)(smem + OFF_ENC);
    __bf16* encpreS = (__bf16*)(smem + OFF_ENCPRE);
    __bf16* hidS    = (__bf16*)(smem + OFF_HID);
    const __bf16* pW2   = (const __bf16*)pW2u;
    const __bf16* pA1b  = (const __bf16*)pA1bu;
    const __bf16* pA1ng = (const __bf16*)pA1ngu;

    const int tid  = threadIdx.x;
    const int base = blockIdx.x * TP;

    // ---- Phase 0a: stage W1, build feature rows, encoding tile, clear res --
    for (int i = tid; i < NV * 7 * 256; i += 128) W1s[i] = W1[i];
    if (tid < ROWS) {
        int p = tid / 7, s = tid % 7, gp = base + p;
        float f[7];
        if (s == 0) {
            f[0] = centers[gp * 3 + 0]; f[1] = centers[gp * 3 + 1]; f[2] = centers[gp * 3 + 2];
            f[3] = normals[gp * 3 + 0]; f[4] = normals[gp * 3 + 1]; f[5] = normals[gp * 3 + 2];
            f[6] = logf(areas[gp]) * 0.1f;
        } else {
            int sn = gp * 6 + (s - 1);
            f[0] = nbrs[sn * 3 + 0] + 1e-6f; f[1] = nbrs[sn * 3 + 1] + 1e-6f; f[2] = nbrs[sn * 3 + 2] + 1e-6f;
            f[3] = nb_normals[sn * 3 + 0] + 1e-6f; f[4] = nb_normals[sn * 3 + 1] + 1e-6f;
            f[5] = nb_normals[sn * 3 + 2] + 1e-6f;
            f[6] = logf(nb_areas[sn]) * 0.1f + 1e-6f;
        }
#pragma unroll
        for (int i = 0; i < 7; ++i) rowsS[tid * 8 + i] = f[i];
        rowsS[tid * 8 + 7] = 0.0f;
    }
    for (int i = tid; i < TP * 192; i += 128) {
        int p = i / 192, c = i % 192;
        float val = (c < 128) ? enc_node[(base + p) * 128 + c] : enc_g[(base + p) * 64 + (c - 128)];
        encS[i] = (__bf16)val;
    }
    if (tid < NV * TP) resS[tid] = 0.0f;
    __syncthreads();

    // ---- Phase 0b: inverse distances (7-dim feature distance) --------------
    if (tid < TP * 6) {
        int p = tid / 6, s = tid % 6;
        const float* c = rowsS + (p * 7) * 8;
        const float* n = rowsS + (p * 7 + 1 + s) * 8;
        float d2 = 0.0f;
#pragma unroll
        for (int i = 0; i < 7; ++i) { float d = c[i] - n[i]; d2 = fmaf(d, d, d2); }
        invS[tid] = 1.0f / sqrtf(d2);
    }
    __syncthreads();

    const int v    = tid >> 5;        // wave id == variable id
    const int lane = tid & 31;
    const int col  = lane & 15;
    const int lo   = (lane < 16) ? 0 : 8;   // fragment K-half offset

    // ---- Phase 1 (VALU): hidden = relu(rows * W1 + b1), K=7 ---------------
    __bf16* hidv = hidS + v * ROWS * 256;
    const float* b1v = b1 + v * 256;
    for (int idx = lane; idx < ROWS * 256; idx += 32) {
        int row = idx >> 8, o = idx & 255;
        float acc = b1v[o];
        const float* w  = W1s + v * 1792 + o;
        const float* rr = rowsS + row * 8;
#pragma unroll
        for (int i = 0; i < 7; ++i) acc = fmaf(rr[i], w[i * 256], acc);
        hidv[idx] = (__bf16)fmaxf(acc, 0.0f);
    }

    // ---- Phase 2 (WMMA): enc_pre = enc(16x192) x A1ng(192x256) + Ab1 ------
    for (int nt = 0; nt < 16; ++nt) {
        float ab = Ab1[v * 256 + nt * 16 + col];
        v8f acc = {ab, ab, ab, ab, ab, ab, ab, ab};
#pragma unroll
        for (int kc = 0; kc < 6; ++kc) {
            const __bf16* ap = encS + (lane & 15) * 192 + kc * 32 + lo;
            v16bf a = load_frag(ap, ap + 16);
            v16bf b = *(const v16bf*)(pA1ng + (((v * 16 + nt) * 6 + kc) * 32 + lane) * 16);
            acc = __builtin_amdgcn_wmma_f32_16x16x32_bf16(false, a, false, b, (short)0, acc, false, false);
        }
#pragma unroll
        for (int r = 0; r < 8; ++r) {
            int p = r + ((lane < 16) ? 0 : 8);
            encpreS[(v * 16 + p) * 256 + nt * 16 + col] = (__bf16)acc[r];
        }
    }

    // ---- Phase 3 (WMMA): basis = hidden(112x256) x W2(256x128) + b2 -------
    __bf16* basv = hidv;   // alias: basis tiles overwrite consumed hidden tiles
    for (int m = 0; m < 7; ++m) {
        v16bf a[8];
#pragma unroll
        for (int kc = 0; kc < 8; ++kc) {
            const __bf16* ap = hidv + (m * 16 + (lane & 15)) * 256 + kc * 32 + lo;
            a[kc] = load_frag(ap, ap + 16);
        }
#pragma unroll
        for (int nt = 0; nt < 8; ++nt) {
            float bb = b2[v * 128 + nt * 16 + col];
            v8f acc = {bb, bb, bb, bb, bb, bb, bb, bb};
            __builtin_prefetch(pW2 + (((v * 8 + nt) * 8) * 32 + lane) * 16, 0, 0);
#pragma unroll
            for (int kc = 0; kc < 8; ++kc) {
                v16bf b = *(const v16bf*)(pW2 + (((v * 8 + nt) * 8 + kc) * 32 + lane) * 16);
                acc = __builtin_amdgcn_wmma_f32_16x16x32_bf16(false, a[kc], false, b, (short)0, acc, false, false);
            }
#pragma unroll
            for (int r = 0; r < 8; ++r) {
                int rg = m * 16 + r + ((lane < 16) ? 0 : 8);
                basv[rg * 128 + nt * 16 + col] = (__bf16)acc[r];
            }
        }
    }

    // ---- Phase 4 (WMMA): h = relu(basis x A1b + enc_pre); out = h . A2 ----
    float a2c[16];
#pragma unroll
    for (int nt = 0; nt < 16; ++nt) a2c[nt] = A2[v * 256 + nt * 16 + col];
    const float ab2v = Ab2[v];

    for (int m = 0; m < 7; ++m) {
        v16bf a[4];
#pragma unroll
        for (int kc = 0; kc < 4; ++kc) {
            const __bf16* ap = basv + (m * 16 + (lane & 15)) * 128 + kc * 32 + lo;
            a[kc] = load_frag(ap, ap + 16);
        }
        float part[8];
#pragma unroll
        for (int r = 0; r < 8; ++r) part[r] = 0.0f;

        for (int nt = 0; nt < 16; ++nt) {
            v8f acc;
#pragma unroll
            for (int r = 0; r < 8; ++r) {           // C init = enc_pre[point]
                int rg = m * 16 + r + ((lane < 16) ? 0 : 8);
                int p = rg / 7;
                acc[r] = (float)encpreS[(v * 16 + p) * 256 + nt * 16 + col];
            }
#pragma unroll
            for (int kc = 0; kc < 4; ++kc) {
                v16bf b = *(const v16bf*)(pA1b + (((v * 16 + nt) * 4 + kc) * 32 + lane) * 16);
                acc = __builtin_amdgcn_wmma_f32_16x16x32_bf16(false, a[kc], false, b, (short)0, acc, false, false);
            }
#pragma unroll
            for (int r = 0; r < 8; ++r) part[r] = fmaf(fmaxf(acc[r], 0.0f), a2c[nt], part[r]);
        }
        // reduce over the 16 lanes that share the same M rows (N across lanes)
#pragma unroll
        for (int r = 0; r < 8; ++r) {
            part[r] += __shfl_xor(part[r], 8, 16);
            part[r] += __shfl_xor(part[r], 4, 16);
            part[r] += __shfl_xor(part[r], 2, 16);
            part[r] += __shfl_xor(part[r], 1, 16);
        }
        if ((lane & 15) == 0) {
#pragma unroll
            for (int r = 0; r < 8; ++r) {
                int rg = m * 16 + r + ((lane < 16) ? 0 : 8);
                int p = rg / 7, s = rg % 7;
                float val = part[r] + ab2v;
                float w;
                if (s == 0) {
                    w = 0.5f;
                } else {
                    float isum = 0.0f;
#pragma unroll
                    for (int q = 0; q < 6; ++q) isum += invS[p * 6 + q];
                    w = 0.5f * invS[p * 6 + (s - 1)] / isum;
                }
                atomicAdd(&resS[v * 16 + p], w * val);  // ds_add_f32
            }
        }
    }
    __syncthreads();

    if (tid < NV * TP) {
        int vv = tid >> 4, p = tid & 15;
        out[(base + p) * NV + vv] = resS[tid];   // output (B, N, V)
    }
}

// ---------------------------------------------------------------------------
extern "C" void kernel_launch(void* const* d_in, const int* in_sizes, int n_in,
                              void* d_out, int out_size, void* d_ws, size_t ws_size,
                              hipStream_t stream) {
    const float* centers    = (const float*)d_in[0];
    const float* enc_g      = (const float*)d_in[1];
    const float* enc_node   = (const float*)d_in[2];
    const float* nbrs       = (const float*)d_in[3];
    const float* normals    = (const float*)d_in[4];
    const float* nb_normals = (const float*)d_in[5];
    const float* areas      = (const float*)d_in[6];
    const float* nb_areas   = (const float*)d_in[7];
    const float* W1  = (const float*)d_in[10];
    const float* b1  = (const float*)d_in[11];
    const float* W2  = (const float*)d_in[12];
    const float* b2  = (const float*)d_in[13];
    const float* A1  = (const float*)d_in[14];
    const float* Ab1 = (const float*)d_in[15];
    const float* A2  = (const float*)d_in[16];
    const float* Ab2 = (const float*)d_in[17];

    unsigned short* pW2   = (unsigned short*)d_ws;
    unsigned short* pA1b  = pW2 + NW2;
    unsigned short* pA1ng = pA1b + NA1B;

    int packTotal = NW2 + NA1B + NA1NG;           // 458752
    pack_weights<<<(packTotal + 255) / 256, 256, 0, stream>>>(W2, A1, pW2, pA1b, pA1ng);

    fused_kernel<<<30000 / TP, 128, LDS_TOTAL, stream>>>(
        centers, enc_g, enc_node, nbrs, normals, nb_normals, areas, nb_areas,
        W1, b1, b2, Ab1, A2, Ab2, pW2, pA1b, pA1ng, (float*)d_out);
}
